// SAModule_51453708206219
// MI455X (gfx1250) — matrix operand order repaired and checked by reading, not compiled, not run on hardware
//
#include <hip/hip_runtime.h>
#include <cstdint>
#include <cstddef>

// ---------------------------------------------------------------------------
// Types for CDNA5 WMMA (wave32)
// ---------------------------------------------------------------------------
typedef __attribute__((ext_vector_type(16))) __bf16        bf16x16;
typedef __attribute__((ext_vector_type(8)))  float         floatx8;
typedef __attribute__((ext_vector_type(4)))  unsigned int  uint32x4;

union Frag16 {           // 16 bf16 = 32 bytes = two 16B chunks
    bf16x16  v;
    uint32x4 q[2];
};

// Problem constants (match reference)
#define BCL   8
#define NPER  4096
#define MPER  1024
#define KNBR  64
#define CIN   64
#define R2    0.04f
#define NCENT (BCL * MPER)          // 8192

__device__ __forceinline__ unsigned short f2bf(float f) {
    unsigned int u = __float_as_uint(f);
    unsigned int r = u + 0x7FFFu + ((u >> 16) & 1u);   // RNE
    return (unsigned short)(r >> 16);
}

// ---------------------------------------------------------------------------
// Kernel 0: convert a weight matrix [din x N] f32 -> bf16 in B-fragment
// swizzled layout: tiles of 32(K) x 16(N); within a tile, 32 lanes x 16
// contiguous elements. lane l: col = l&15, h = l>>4, elem e -> K = (e<8) ? 8h+e : 16+8h+(e-8).
// Rows k >= din are zero-padded.
// ---------------------------------------------------------------------------
__global__ void wswz_kernel(const float* __restrict__ W, unsigned short* __restrict__ dst,
                            int din, int N, int total) {
    int NTt = N >> 4;
    for (int e = blockIdx.x * blockDim.x + threadIdx.x; e < total;
         e += gridDim.x * blockDim.x) {
        int tile   = e >> 9;          // /512 elements per tile
        int within = e & 511;
        int lane   = within >> 4;
        int el     = within & 15;
        int nt     = tile % NTt;
        int kt     = tile / NTt;
        int col    = nt * 16 + (lane & 15);
        int k      = kt * 32 + ((el < 8) ? ((lane >> 4) * 8 + el) : (16 + (lane >> 4) * 8 + (el - 8)));
        float v    = (k < din) ? W[(size_t)k * N + col] : 0.0f;
        dst[e]     = f2bf(v);
    }
}

// ---------------------------------------------------------------------------
// Kernel 1: farthest point sampling, one block per cloud. Deterministic
// start at index 0 (matches reference).
// ---------------------------------------------------------------------------
__global__ __launch_bounds__(256) void fps_kernel(const float* __restrict__ pos,
                                                  int* __restrict__ fps_idx) {
    __shared__ float sx[NPER], sy[NPER], sz[NPER];
    __shared__ float sval[256];
    __shared__ int   sidx[256];
    __shared__ int   scur;
    const int b = blockIdx.x, tid = threadIdx.x;
    const int base = b * NPER;
    for (int i = tid; i < NPER; i += 256) {
        sx[i] = pos[(size_t)(base + i) * 3 + 0];
        sy[i] = pos[(size_t)(base + i) * 3 + 1];
        sz[i] = pos[(size_t)(base + i) * 3 + 2];
    }
    float d[16];
#pragma unroll
    for (int i = 0; i < 16; ++i) d[i] = 3.0e38f;
    if (tid == 0) scur = 0;
    __syncthreads();

    for (int m = 0; m < MPER; ++m) {
        const int cur = scur;
        if (tid == 0) fps_idx[b * MPER + m] = cur;
        const float cx = sx[cur], cy = sy[cur], cz = sz[cur];
        float bestv = -1.0f; int bestj = 0;
#pragma unroll
        for (int i = 0; i < 16; ++i) {
            int j = i * 256 + tid;
            float dx = sx[j] - cx, dy = sy[j] - cy, dz = sz[j] - cz;
            float nd = fminf(d[i], dx * dx + dy * dy + dz * dz);
            d[i] = nd;
            if (nd > bestv) { bestv = nd; bestj = j; }
        }
        sval[tid] = bestv; sidx[tid] = bestj;
        __syncthreads();
        for (int s = 128; s > 0; s >>= 1) {
            if (tid < s && sval[tid + s] > sval[tid]) {
                sval[tid] = sval[tid + s]; sidx[tid] = sidx[tid + s];
            }
            __syncthreads();
        }
        if (tid == 0) scur = sidx[0];
        __syncthreads();
    }
}

// ---------------------------------------------------------------------------
// Kernel 2: 64-NN per centroid + radius validity mask. One block per centroid.
// Also emits pos_out (centroid positions) and batch_out.
// ---------------------------------------------------------------------------
__global__ __launch_bounds__(256) void knn_kernel(const float* __restrict__ pos,
                                                  const int* __restrict__ fps_idx,
                                                  int* __restrict__ nbr,
                                                  unsigned long long* __restrict__ maskArr,
                                                  float* __restrict__ pos_out,
                                                  int* __restrict__ batch_out) {
    __shared__ float sd[NPER];
    __shared__ float sval[256];
    __shared__ int   sidx[256];
    const int c = blockIdx.x, tid = threadIdx.x;
    const int b = c >> 10;
    const int base = b * NPER;
    const int qi = fps_idx[c];
    const float qx = pos[(size_t)(base + qi) * 3 + 0];
    const float qy = pos[(size_t)(base + qi) * 3 + 1];
    const float qz = pos[(size_t)(base + qi) * 3 + 2];
    if (tid == 0) {
        pos_out[(size_t)c * 3 + 0] = qx;
        pos_out[(size_t)c * 3 + 1] = qy;
        pos_out[(size_t)c * 3 + 2] = qz;
        batch_out[c] = b;
    }
    for (int j = tid; j < NPER; j += 256) {
        float dx = pos[(size_t)(base + j) * 3 + 0] - qx;
        float dy = pos[(size_t)(base + j) * 3 + 1] - qy;
        float dz = pos[(size_t)(base + j) * 3 + 2] - qz;
        sd[j] = dx * dx + dy * dy + dz * dz;
    }
    __syncthreads();
    unsigned long long mask = 0ull;
    for (int k = 0; k < KNBR; ++k) {
        float bv = 3.0e38f; int bj = 0;
#pragma unroll
        for (int i = 0; i < 16; ++i) {
            int j = i * 256 + tid;
            float v = sd[j];
            if (v < bv) { bv = v; bj = j; }
        }
        sval[tid] = bv; sidx[tid] = bj;
        __syncthreads();
        for (int s = 128; s > 0; s >>= 1) {
            if (tid < s && sval[tid + s] < sval[tid]) {
                sval[tid] = sval[tid + s]; sidx[tid] = sidx[tid + s];
            }
            __syncthreads();
        }
        const int   j  = sidx[0];
        const float dv = sval[0];
        if (tid == 0) { nbr[(size_t)c * KNBR + k] = j; sd[j] = 3.0e38f; }
        if (dv <= R2) mask |= (1ull << k);
        __syncthreads();
    }
    if (tid == 0) maskArr[c] = mask;
}

// ---------------------------------------------------------------------------
// Core GEMM micro-tile: one wave computes a full 64(M) x 16(N) column strip.
// B fragments (KT <= 4) are preloaded into registers (one clause of
// global_load_b128 pairs, L2-resident), then KT x 4 WMMAs run back-to-back
// fed only by LDS A-fragment loads. acc[mt] RAW chains are 4 WMMAs apart.
// ---------------------------------------------------------------------------
template <int KT, int NTt, int STRIDE>
static __device__ __forceinline__ void
mma_colstrip(const unsigned short* __restrict__ ldsA,
             const unsigned short* __restrict__ wswz,
             int nt, int lane, floatx8 acc[4]) {
    Frag16 bfr[KT];
    const unsigned short* bbase = wswz + (size_t)nt * 512 + (size_t)lane * 16;
#pragma unroll
    for (int kt = 0; kt < KT; ++kt) {
        const unsigned short* bp = bbase + (size_t)kt * NTt * 512;
        bfr[kt].q[0] = *reinterpret_cast<const uint32x4*>(bp);
        bfr[kt].q[1] = *reinterpret_cast<const uint32x4*>(bp + 8);
    }
    const int row   = lane & 15;
    const int khalf = (lane < 16) ? 0 : 8;
#pragma unroll
    for (int kt = 0; kt < KT; ++kt) {
#pragma unroll
        for (int mt = 0; mt < 4; ++mt) {
            Frag16 a;
            const unsigned short* ap =
                ldsA + (size_t)(mt * 16 + row) * STRIDE + kt * 32 + khalf;
            a.q[0] = *reinterpret_cast<const uint32x4*>(ap);
            a.q[1] = *reinterpret_cast<const uint32x4*>(ap + 16);
            acc[mt] = __builtin_amdgcn_wmma_f32_16x16x32_bf16(
                /*neg_a=*/false, a.v, /*neg_b=*/false, bfr[kt].v,
                /*c_mod=*/(short)0, acc[mt], /*reuse_a=*/false, /*reuse_b=*/false);
        }
    }
}

// ---------------------------------------------------------------------------
// Kernel 3: fused gather + 3-layer MLP (bf16 WMMA) + masked max aggregation.
// One block per centroid: 128 threads = 4 waves. Waves split the N (output
// channel) dimension; each wave computes all 64 neighbor rows for its
// columns, so layer-3 max aggregation completes entirely in-wave.
// ---------------------------------------------------------------------------
__global__ __launch_bounds__(128)
void sa_mlp_kernel(const float* __restrict__ x, const float* __restrict__ pos,
                   const int* __restrict__ fps_idx, const int* __restrict__ nbr,
                   const unsigned long long* __restrict__ maskArr,
                   const unsigned short* __restrict__ w1swz,
                   const unsigned short* __restrict__ w2swz,
                   const unsigned short* __restrict__ w3swz,
                   const float* __restrict__ b1, const float* __restrict__ b2,
                   const float* __restrict__ b3, float* __restrict__ out) {
    __shared__ __align__(16) unsigned short sA [64 * 96];    // 12 KB, K padded 67->96
    __shared__ __align__(16) unsigned short sH1[64 * 128];   // 16 KB
    __shared__ __align__(16) unsigned short sH2[64 * 128];   // 16 KB

    const int c   = blockIdx.x, tid = threadIdx.x;
    const int b   = c >> 10;
    const int base = b * NPER;
    const unsigned long long mask = maskArr[c];
    const int qi = fps_idx[c];
    const float q0 = pos[(size_t)(base + qi) * 3 + 0];
    const float q1 = pos[(size_t)(base + qi) * 3 + 1];
    const float q2 = pos[(size_t)(base + qi) * 3 + 2];

    // ---- gather features [64 x 96] bf16 into LDS: [x_j (64) | rel (3) | pad] ----
    for (int e = tid; e < 64 * 96; e += 128) {
        int r  = e / 96;
        int cc = e - r * 96;
        int j  = nbr[(size_t)c * KNBR + r];
        float v;
        if (cc < CIN)        v = x[(size_t)(base + j) * CIN + cc];
        else if (cc == 64)   v = pos[(size_t)(base + j) * 3 + 0] - q0;
        else if (cc == 65)   v = pos[(size_t)(base + j) * 3 + 1] - q1;
        else if (cc == 66)   v = pos[(size_t)(base + j) * 3 + 2] - q2;
        else                 v = 0.0f;
        sA[e] = f2bf(v);
    }
    __syncthreads();

    const int lane = tid & 31;
    const int wave = tid >> 5;
    const int rsel = (lane >= 16) ? 8 : 0;     // row offset of C-fragment half

    // ---- layer 1: [64x96] @ [96x128] -> relu -> sH1 (wave owns 2 N-tiles) ----
#pragma unroll
    for (int t = 0; t < 2; ++t) {
        const int nt = wave * 2 + t;
        floatx8 acc[4] = {floatx8{}, floatx8{}, floatx8{}, floatx8{}};
        mma_colstrip<3, 8, 96>(sA, w1swz, nt, lane, acc);
        const int col   = nt * 16 + (lane & 15);
        const float bias = b1[col];
#pragma unroll
        for (int mt = 0; mt < 4; ++mt) {
            const int rbase = mt * 16 + rsel;
#pragma unroll
            for (int r = 0; r < 8; ++r) {
                float v = fmaxf(acc[mt][r] + bias, 0.0f);
                sH1[(size_t)(rbase + r) * 128 + col] = f2bf(v);
            }
        }
    }
    __syncthreads();

    // ---- layer 2: [64x128] @ [128x128] -> relu -> sH2 (wave owns 2 N-tiles) ----
#pragma unroll
    for (int t = 0; t < 2; ++t) {
        const int nt = wave * 2 + t;
        floatx8 acc[4] = {floatx8{}, floatx8{}, floatx8{}, floatx8{}};
        mma_colstrip<4, 8, 128>(sH1, w2swz, nt, lane, acc);
        const int col   = nt * 16 + (lane & 15);
        const float bias = b2[col];
#pragma unroll
        for (int mt = 0; mt < 4; ++mt) {
            const int rbase = mt * 16 + rsel;
#pragma unroll
            for (int r = 0; r < 8; ++r) {
                float v = fmaxf(acc[mt][r] + bias, 0.0f);
                sH2[(size_t)(rbase + r) * 128 + col] = f2bf(v);
            }
        }
    }
    __syncthreads();

    // ---- layer 3: [64x128] @ [128x256] -> relu -> masked max over all 64 rows.
    //      Wave owns 4 N-tiles (64 cols); the row-max is entirely in-wave. ----
#pragma unroll
    for (int t = 0; t < 4; ++t) {
        const int nt = wave * 4 + t;
        floatx8 acc[4] = {floatx8{}, floatx8{}, floatx8{}, floatx8{}};
        mma_colstrip<4, 16, 128>(sH2, w3swz, nt, lane, acc);
        const int col   = nt * 16 + (lane & 15);
        const float bias = b3[col];
        float mx = -1.0e30f;
#pragma unroll
        for (int mt = 0; mt < 4; ++mt) {
            const int rbase = mt * 16 + rsel;
#pragma unroll
            for (int r = 0; r < 8; ++r) {
                float v  = fmaxf(acc[mt][r] + bias, 0.0f);
                bool  ok = ((mask >> (unsigned)(rbase + r)) & 1ull) != 0ull;
                mx = fmaxf(mx, ok ? v : -1.0e30f);
            }
        }
        mx = fmaxf(mx, __shfl_xor(mx, 16, 32));   // merge row halves (wave32)
        if (lane < 16)
            out[(size_t)c * 256 + col] = (mask != 0ull) ? mx : 0.0f;
    }
}

// ---------------------------------------------------------------------------
// Host launcher
// ---------------------------------------------------------------------------
extern "C" void kernel_launch(void* const* d_in, const int* in_sizes, int n_in,
                              void* d_out, int out_size, void* d_ws, size_t ws_size,
                              hipStream_t stream) {
    (void)in_sizes; (void)n_in; (void)out_size; (void)ws_size;

    const float* x   = (const float*)d_in[0];
    const float* pos = (const float*)d_in[1];
    /* d_in[2] = batch, unused (clouds are contiguous by construction) */
    const float* W1  = (const float*)d_in[3];
    const float* b1  = (const float*)d_in[4];
    const float* W2  = (const float*)d_in[5];
    const float* b2  = (const float*)d_in[6];
    const float* W3  = (const float*)d_in[7];
    const float* b3  = (const float*)d_in[8];

    // outputs, concatenated flat: out [8192x256] f32 | pos_out [8192x3] f32 | batch_out [8192] i32
    float* out       = (float*)d_out;
    float* pos_out   = out + (size_t)NCENT * 256;
    int*   batch_out = (int*)(out + (size_t)NCENT * 256 + (size_t)NCENT * 3);

    // workspace carve-up (~2.3 MB)
    unsigned short*     w1swz   = (unsigned short*)d_ws;            // 96*128  = 12288
    unsigned short*     w2swz   = w1swz + 96 * 128;                 // 128*128 = 16384
    unsigned short*     w3swz   = w2swz + 128 * 128;                // 128*256 = 32768
    int*                fps_idx = (int*)(w3swz + 128 * 256);        // 8192 ints
    int*                nbr     = fps_idx + NCENT;                  // 8192*64 ints
    unsigned long long* maskArr = (unsigned long long*)(nbr + (size_t)NCENT * KNBR);

    // 0) weight conversion to bf16 swizzled B-fragment layout
    wswz_kernel<<<48,  256, 0, stream>>>(W1, w1swz,  67, 128,  96 * 128);
    wswz_kernel<<<64,  256, 0, stream>>>(W2, w2swz, 128, 128, 128 * 128);
    wswz_kernel<<<128, 256, 0, stream>>>(W3, w3swz, 128, 256, 128 * 256);

    // 1) farthest point sampling (one block per cloud)
    fps_kernel<<<BCL, 256, 0, stream>>>(pos, fps_idx);

    // 2) 64-NN + radius mask + pos_out/batch_out (one block per centroid)
    knn_kernel<<<NCENT, 256, 0, stream>>>(pos, fps_idx, nbr, maskArr, pos_out, batch_out);

    // 3) fused gather + MLP (bf16 WMMA, N-split waves) + masked max aggregation
    sa_mlp_kernel<<<NCENT, 128, 0, stream>>>(x, pos, fps_idx, nbr, maskArr,
                                             w1swz, w2swz, w3swz, b1, b2, b3, out);
}
